// MlpWithCudaKernel_12171937317146
// MI455X (gfx1250) — 8-bit matrix operand order repaired and checked by reading, compiled, not run on hardware
//
#include <hip/hip_runtime.h>
#include <math.h>


typedef int v8i __attribute__((ext_vector_type(8)));

static constexpr int T_TOK = 8192;
static constexpr int D_DIM = 5120;
static constexpr int H_DIM = 13824;

__global__ void pack_i32_to_u8(const int* __restrict__ src,
                               unsigned char* __restrict__ dst, size_t n) {
  size_t i = ((size_t)blockIdx.x * blockDim.x + threadIdx.x) * 4;
  size_t stride = (size_t)gridDim.x * blockDim.x * 4;
  for (; i < n; i += stride) {
    int4 v = *(const int4*)(src + i);
    uchar4 o;
    o.x = (unsigned char)v.x; o.y = (unsigned char)v.y;
    o.z = (unsigned char)v.z; o.w = (unsigned char)v.w;
    *(uchar4*)(dst + i) = o;
  }
}

__global__ __launch_bounds__(256)
void quant_x_kernel(const float* __restrict__ x, signed char* __restrict__ q,
                    float* __restrict__ s_out, float* __restrict__ sum_out,
                    int cols) {
  __shared__ float sm[8];
  __shared__ float sbc;
  const int row = blockIdx.x;
  const float* xr = x + (size_t)row * cols;

  float amax = 0.0f;
  for (int c = threadIdx.x; c < cols; c += 256) amax = fmaxf(amax, fabsf(xr[c]));
#pragma unroll
  for (int o = 16; o; o >>= 1) amax = fmaxf(amax, __shfl_xor(amax, o, 32));
  if ((threadIdx.x & 31) == 0) sm[threadIdx.x >> 5] = amax;
  __syncthreads();
  if (threadIdx.x == 0) {
    float m = sm[0];
#pragma unroll
    for (int w = 1; w < 8; w++) m = fmaxf(m, sm[w]);
    sbc = fmaxf(m / 127.0f, 1e-8f);
  }
  __syncthreads();
  const float s = sbc;
  const float inv = 1.0f / s;

  float sum = 0.0f;
  signed char* qr = q + (size_t)row * cols;
  for (int c = threadIdx.x; c < cols; c += 256) {
    float v = rintf(xr[c] * inv);
    v = fminf(fmaxf(v, -127.0f), 127.0f);
    qr[c] = (signed char)(int)v;
    sum += v;
  }
#pragma unroll
  for (int o = 16; o; o >>= 1) sum += __shfl_xor(sum, o, 32);
  __syncthreads();
  if ((threadIdx.x & 31) == 0) sm[threadIdx.x >> 5] = sum;
  __syncthreads();
  if (threadIdx.x == 0) {
    float t = 0.0f;
#pragma unroll
    for (int w = 0; w < 8; w++) t += sm[w];
    s_out[row] = s;
    sum_out[row] = t;
  }
}

__global__ __launch_bounds__(256)
void quant_g_kernel(const float* __restrict__ g,
                    const unsigned int* __restrict__ amax_bits,
                    signed char* __restrict__ q, float* __restrict__ s_out,
                    float* __restrict__ sum_out, int cols) {
  __shared__ float sm[8];
  const int row = blockIdx.x;
  const float amax = __uint_as_float(amax_bits[row]);
  const float s = fmaxf(amax / 127.0f, 1e-8f);
  const float inv = 1.0f / s;
  const float* gr = g + (size_t)row * cols;
  signed char* qr = q + (size_t)row * cols;

  float sum = 0.0f;
  for (int c = threadIdx.x; c < cols; c += 256) {
    float v = rintf(gr[c] * inv);
    v = fminf(fmaxf(v, -127.0f), 127.0f);
    qr[c] = (signed char)(int)v;
    sum += v;
  }
#pragma unroll
  for (int o = 16; o; o >>= 1) sum += __shfl_xor(sum, o, 32);
  if ((threadIdx.x & 31) == 0) sm[threadIdx.x >> 5] = sum;
  __syncthreads();
  if (threadIdx.x == 0) {
    float t = 0.0f;
#pragma unroll
    for (int w = 0; w < 8; w++) t += sm[w];
    s_out[row] = s;
    sum_out[row] = t;
  }
}

template <bool FUSE_GELU>
__global__ __launch_bounds__(256)
void gemm_w8a8(const signed char* __restrict__ A, const unsigned char* __restrict__ B,
               int K, int N,
               const float* __restrict__ s_row, const float* __restrict__ sum_row,
               const float* __restrict__ s_col, const float* __restrict__ zp_col,
               const float* __restrict__ bias_col,
               float* __restrict__ Out, unsigned int* __restrict__ amax_bits) {
  const int lane = threadIdx.x & 31;
  const int wave = threadIdx.x >> 5;
  const int wm = wave >> 2;
  const int wn = wave & 3;
  const int m0 = blockIdx.x * 128 + wm * 64;
  const int n0 = blockIdx.y * 256 + wn * 64;
  const int lr = lane & 15;
  const int hi = lane >> 4;

  const signed char* ap[4];
  const unsigned char* bp[4];
#pragma unroll
  for (int i = 0; i < 4; i++)
    ap[i] = A + (size_t)(m0 + 16 * i + lr) * K + hi * 8;
#pragma unroll
  for (int j = 0; j < 4; j++)
    bp[j] = B + (size_t)(n0 + 16 * j + lr) * K + hi * 8;

  v8i acc[4][4] = {};

  for (int k = 0; k < K; k += 64) {
    v8i a[4], b[4];
#pragma unroll
    for (int i = 0; i < 4; i++) {
      int2 x0 = *(const int2*)(ap[i] + k);
      int2 x1 = *(const int2*)(ap[i] + k + 16);
      int2 x2 = *(const int2*)(ap[i] + k + 32);
      int2 x3 = *(const int2*)(ap[i] + k + 48);
      v8i t;
      t[0] = x0.x; t[1] = x0.y; t[2] = x1.x; t[3] = x1.y;
      t[4] = x2.x; t[5] = x2.y; t[6] = x3.x; t[7] = x3.y;
      a[i] = t;
    }
#pragma unroll
    for (int j = 0; j < 4; j++) {
      int2 y0 = *(const int2*)(bp[j] + k);
      int2 y1 = *(const int2*)(bp[j] + k + 16);
      int2 y2 = *(const int2*)(bp[j] + k + 32);
      int2 y3 = *(const int2*)(bp[j] + k + 48);
      v8i t;
      t[0] = y0.x; t[1] = y0.y; t[2] = y1.x; t[3] = y1.y;
      t[4] = y2.x; t[5] = y2.y; t[6] = y3.x; t[7] = y3.y;
      b[j] = t;
    }
#pragma unroll
    for (int i = 0; i < 4; i++)
#pragma unroll
      for (int j = 0; j < 4; j++)
        acc[i][j] = __builtin_amdgcn_wmma_i32_16x16x64_iu8(
             true, a[i],  false, b[j], acc[i][j],
             false,  false);
  }

#pragma unroll
  for (int j = 0; j < 4; j++) {
    const int col = n0 + 16 * j + lr;
    const float sw = s_col[col];
    const float zp = zp_col[col];
    const float bs = bias_col[col];
#pragma unroll
    for (int i = 0; i < 4; i++) {
#pragma unroll
      for (int r = 0; r < 8; r++) {
        const int row = m0 + 16 * i + r + hi * 8;
        const float sx = s_row[row];
        const float sq = sum_row[row];
        float h = sx * sw * ((float)acc[i][j][r] - zp * sq) + bs;
        if (FUSE_GELU) {
          float g = 0.5f * h * (1.0f + erff(h * 0.70710678118654752f));
          Out[(size_t)row * N + col] = g;
          float m = fabsf(g);
#pragma unroll
          for (int o = 8; o; o >>= 1) m = fmaxf(m, __shfl_xor(m, o, 16));
          if (lr == 0)
            atomicMax(amax_bits + row, __float_as_uint(m));
        } else {
          Out[(size_t)row * N + col] = h;
        }
      }
    }
  }
}

extern "C" void kernel_launch(void* const* d_in, const int* in_sizes, int n_in,
                              void* d_out, int out_size, void* d_ws, size_t ws_size,
                              hipStream_t stream) {
  const float* x    = (const float*)d_in[0];
  const int*   w1q  = (const int*)d_in[1];
  const float* w1s  = (const float*)d_in[2];
  const float* w1zp = (const float*)d_in[3];
  const float* b1   = (const float*)d_in[4];
  const int*   w2q  = (const int*)d_in[5];
  const float* w2s  = (const float*)d_in[6];
  const float* w2zp = (const float*)d_in[7];
  const float* b2   = (const float*)d_in[8];
  float* out = (float*)d_out;

  const size_t T = T_TOK, D = D_DIM, H = H_DIM;

  char* ws = (char*)d_ws;
  size_t off = 0;
  auto take = [&](size_t bytes) -> char* {
    char* p = ws + off;
    off = (off + bytes + 255) & ~(size_t)255;
    return p;
  };
  unsigned char* w1p   = (unsigned char*)take(H * D);
  unsigned char* w2p   = (unsigned char*)take(D * H);
  signed char*   q1    = (signed char*)take(T * D);
  float*         s1    = (float*)take(T * sizeof(float));
  float*         sum1  = (float*)take(T * sizeof(float));
  float*         gbuf  = (float*)take(T * H * sizeof(float));
  unsigned int*  amax2 = (unsigned int*)take(T * sizeof(unsigned int));
  signed char*   q2    = (signed char*)take(T * H);
  float*         s2    = (float*)take(T * sizeof(float));
  float*         sum2  = (float*)take(T * sizeof(float));

  pack_i32_to_u8<<<4096, 256, 0, stream>>>(w1q, w1p, H * D);
  pack_i32_to_u8<<<4096, 256, 0, stream>>>(w2q, w2p, D * H);

  quant_x_kernel<<<(int)T, 256, 0, stream>>>(x, q1, s1, sum1, (int)D);

  hipMemsetAsync(amax2, 0, T * sizeof(unsigned int), stream);

  dim3 g1((unsigned)(T / 128), (unsigned)(H / 256));
  gemm_w8a8<true><<<g1, 256, 0, stream>>>(q1, w1p, (int)D, (int)H,
                                          s1, sum1, w1s, w1zp, b1, gbuf, amax2);

  quant_g_kernel<<<(int)T, 256, 0, stream>>>(gbuf, amax2, q2, s2, sum2, (int)H);

  dim3 g2((unsigned)(T / 128), (unsigned)(D / 256));
  gemm_w8a8<false><<<g2, 256, 0, stream>>>(q2, w2p, (int)H, (int)D,
                                           s2, sum2, w2s, w2zp, b2, out, nullptr);
}
